// QuantumAttention_2027224564544
// MI455X (gfx1250) — matrix operand order repaired and checked by reading, not compiled, not run on hardware
//
#include <hip/hip_runtime.h>
#include <hip/hip_bf16.h>

// ---------------------------------------------------------------------------
// CDNA5 / gfx1250 fused "quantum attention".
// All GEMMs via v_wmma_f32_16x16x32_f16 (wave32). f16 storage, f32 accum.
// Attention computes S^T = K @ qm^T so softmax reduces over per-lane regs
// (2 shuffles per row-tile) and every memory access is a b128 transaction.
// ---------------------------------------------------------------------------

typedef __attribute__((ext_vector_type(16))) _Float16 v16h;
typedef __attribute__((ext_vector_type(8)))  float    v8f;

#define BQA 2
#define SQA 2048
#define DQA 512
#define HQA 8
#define QQA 64
#define HDQA 64
#define ROWS (BQA * SQA)   // 4096

union U16 { uint4 u[2]; v16h h; };
union U8  { uint4 u; _Float16 h[8]; };

__device__ __forceinline__ v8f wmma_f16(v16h a, v16h b, v8f c) {
  return __builtin_amdgcn_wmma_f32_16x16x32_f16(
      /*neg_a=*/false, a, /*neg_b=*/false, b,
      /*c_mod=*/(short)0, c, /*reuse_a=*/false, /*reuse_b=*/false);
}

// A fragment: 16x32 f16, row-major source, row stride ld (elements).
// Per lane: two contiguous 8-half runs -> two b128 loads.
__device__ __forceinline__ v16h load_a16_vec(const _Float16* src, int ld) {
  const int lane = threadIdx.x & 31;
  const int row  = lane & 15;
  const int kh   = (lane >> 4) * 8;
  const _Float16* p = src + row * ld + kh;
  U16 u;
  u.u[0] = *(const uint4*)(p);        // K = kh .. kh+7
  u.u[1] = *(const uint4*)(p + 16);   // K = kh+16 .. kh+23
  return u.h;
}

// B fragment (32x16): column n of B must be K-contiguous in memory.
__device__ __forceinline__ v16h load_bt_vec(const _Float16* col0, int ldn) {
  const int lane = threadIdx.x & 31;
  const int n    = lane & 15;
  const int kh   = (lane >> 4) * 8;
  const _Float16* p = col0 + n * ldn + kh;
  U16 u;
  u.u[0] = *(const uint4*)(p);        // K = kh .. kh+7
  u.u[1] = *(const uint4*)(p + 16);   // K = kh+16 .. kh+23
  return u.h;
}

// C/D fragment store (16x16 f32 -> f16), row-major, row stride ld.
__device__ __forceinline__ void store_c_f16(_Float16* dst, int ld, v8f c) {
  const int lane = threadIdx.x & 31;
  const int col  = lane & 15;
  const int rb   = (lane >> 4) * 8;
#pragma unroll
  for (int v = 0; v < 8; ++v) dst[(rb + v) * ld + col] = (_Float16)c[v];
}

// C/D fragment store, column-major target (element (r,n) at dst00[n*ldc + r]).
// Per lane the 8 rows are contiguous -> one b128 store.
__device__ __forceinline__ void store_c_f16_col(_Float16* dst00, int ldc, v8f c) {
  const int lane = threadIdx.x & 31;
  const int col  = lane & 15;
  const int rb   = (lane >> 4) * 8;
  U8 u;
#pragma unroll
  for (int v = 0; v < 8; ++v) u.h[v] = (_Float16)c[v];
  *(uint4*)(dst00 + col * ldc + rb) = u.u;
}

// ---------------------------------------------------------------------------
// Prep kernels (tiny, run once per launch).
// ---------------------------------------------------------------------------

__global__ void cvt_f16_kernel(const float* __restrict__ src,
                               _Float16* __restrict__ dst, int n) {
  int i = blockIdx.x * blockDim.x + threadIdx.x;
  if (i < n) dst[i] = (_Float16)src[i];
}

// dst[c*R + r] = (f16)src[r*C + c]   (transpose RxC -> CxR)
__global__ void cvt_t_kernel(const float* __restrict__ src,
                             _Float16* __restrict__ dst, int R, int C) {
  int i = blockIdx.x * blockDim.x + threadIdx.x;
  if (i >= R * C) return;
  int r = i / C, c = i % C;
  dst[c * R + r] = (_Float16)src[i];
}

// CT(64x64) = (A(64x64) @ B(64x64))^T, f32 in, f16 out. 4096 threads.
__global__ void gemm64_t_kernel(const float* __restrict__ A,
                                const float* __restrict__ B,
                                _Float16* __restrict__ CT) {
  int i = blockIdx.x * blockDim.x + threadIdx.x;
  if (i >= 64 * 64) return;
  int r = i >> 6, c = i & 63;
  float s = 0.f;
#pragma unroll 8
  for (int k = 0; k < 64; ++k) s += A[r * 64 + k] * B[k * 64 + c];
  CT[c * 64 + r] = (_Float16)s;
}

// W_effT[h,e,d] = supw[h,d] * interf[h,d,e] * supw[h,e]
__global__ void fold_interf_t_kernel(const float* __restrict__ interf,
                                     const float* __restrict__ supw,
                                     _Float16* __restrict__ out) {
  int i = blockIdx.x * blockDim.x + threadIdx.x;
  if (i >= HQA * HDQA * HDQA) return;
  int h = i >> 12, d = (i >> 6) & 63, e = i & 63;
  out[(h << 12) + (e << 6) + d] =
      (_Float16)(interf[i] * supw[(h << 6) + d] * supw[(h << 6) + e]);
}

// w_out_effT[e,d] = supw.flat[d] * w_out[d,e]
__global__ void fold_wout_t_kernel(const float* __restrict__ wout,
                                   const float* __restrict__ supw,
                                   _Float16* __restrict__ out) {
  int i = blockIdx.x * blockDim.x + threadIdx.x;
  if (i >= DQA * DQA) return;
  int d = i >> 9, e = i & 511;
  out[e * DQA + d] = (_Float16)(wout[i] * supw[d]);
}

// ---------------------------------------------------------------------------
// QNL: out = (|xe@GR|^2 + |xe@GI|^2) @ meas, xe = x@enc.  16 rows per block.
// TRV=true stores the result transposed (vt[d][row]) for the V layer.
// ---------------------------------------------------------------------------
template <bool TRV>
__global__ void __launch_bounds__(128) qnl_kernel(
    const _Float16* __restrict__ xh,     // 4096x512 row-major
    const _Float16* __restrict__ encT,   // 64x512  [n][k]
    const _Float16* __restrict__ GRT,    // 64x64   [n][k]  ((gr@ent)^T)
    const _Float16* __restrict__ GIT,    // 64x64   [n][k]
    const _Float16* __restrict__ measT,  // 512x64  [n][k]
    _Float16* __restrict__ out)          // 4096x512 or transposed 512x4096
{
  __shared__ _Float16 xeS[16 * 64];
  __shared__ _Float16 probS[16 * 64];
  const int w    = threadIdx.x >> 5;
  const int row0 = blockIdx.x * 16;

  // xe tile (16x16 per wave), K = 512
  {
    v8f c = {};
#pragma unroll 4
    for (int ks = 0; ks < 16; ++ks) {
      v16h a = load_a16_vec(xh + row0 * DQA + ks * 32, DQA);
      v16h b = load_bt_vec(encT + (w * 16) * DQA + ks * 32, DQA);
      c = wmma_f16(a, b, c);
    }
    store_c_f16(xeS + w * 16, 64, c);
  }
  __syncthreads();

  // er/ei/prob tile (16x16 per wave), K = 64
  {
    v16h a0 = load_a16_vec(xeS, 64);
    v16h a1 = load_a16_vec(xeS + 32, 64);
    v8f cr = {}, ci = {};
    cr = wmma_f16(a0, load_bt_vec(GRT + (w * 16) * 64, 64), cr);
    cr = wmma_f16(a1, load_bt_vec(GRT + (w * 16) * 64 + 32, 64), cr);
    ci = wmma_f16(a0, load_bt_vec(GIT + (w * 16) * 64, 64), ci);
    ci = wmma_f16(a1, load_bt_vec(GIT + (w * 16) * 64 + 32, 64), ci);
    v8f p;
#pragma unroll
    for (int v = 0; v < 8; ++v) p[v] = cr[v] * cr[v] + ci[v] * ci[v];
    store_c_f16(probS + w * 16, 64, p);
  }
  __syncthreads();

  // out = prob @ meas, 8 col-tiles per wave, K = 64
  {
    v16h a0 = load_a16_vec(probS, 64);
    v16h a1 = load_a16_vec(probS + 32, 64);
#pragma unroll 2
    for (int j = 0; j < 8; ++j) {
      const int ncol = w * 128 + j * 16;
      v8f c = {};
      c = wmma_f16(a0, load_bt_vec(measT + ncol * 64, 64), c);
      c = wmma_f16(a1, load_bt_vec(measT + ncol * 64 + 32, 64), c);
      if (TRV)
        store_c_f16_col(out + ncol * ROWS + row0, ROWS, c);  // b128 stores
      else
        store_c_f16(out + row0 * DQA + ncol, DQA, c);
    }
  }
}

// ---------------------------------------------------------------------------
// Flash attention per (b,h), transposed-score formulation.
//   qm^T = W_eff^T @ q^T            (pre-scaled by log2(e)/sqrt(HD))
//   S^T  = K_tile @ qm^T            (lane = query row, regs = keys)
//   O^T  = V^T @ P^T                (lane = query row, regs = hd)
// Block = 4 waves; each wave owns 32 query rows (2 row-tiles). Per 32-key
// iteration: 16 WMMA, 16 global b128, 8 LDS ops, 4 shuffles.
// ---------------------------------------------------------------------------
__global__ void __launch_bounds__(128) attn_kernel(
    const _Float16* __restrict__ qh,     // 4096x512 row-major
    const _Float16* __restrict__ kh,     // 4096x512 row-major
    const _Float16* __restrict__ vt,     // 512x4096 transposed: vt[d][row]
    const _Float16* __restrict__ WeffT,  // 8 x 64 x 64  [h][e][d]
    _Float16* __restrict__ aoh)          // 4096x512
{
  __shared__ _Float16 qmS[4][32 * 64];   // qm row-major [qrow][e], per wave
  __shared__ _Float16 pS[4][32 * 32];    // P row-major [qrow][key], per wave
  const int w    = threadIdx.x >> 5;
  const int lane = threadIdx.x & 31;
  const int bh   = blockIdx.y;
  const int b    = bh >> 3, h = bh & 7;
  const _Float16* qp  = qh + (b * SQA) * DQA + h * HDQA;
  const _Float16* kp  = kh + (b * SQA) * DQA + h * HDQA;
  const _Float16* vtp = vt + (h * HDQA) * ROWS + b * SQA;
  const int q0 = blockIdx.x * 128 + w * 32;   // first of 32 query rows

  const float SC = 0.18033688011112042f;  // log2(e) / sqrt(64)

  // qm^T = W_eff^T @ q^T : A = WeffT rows (e), B = q columns (d-contiguous).
  // C tile (e x qrow) stored column-major -> qm row-major in LDS, b128.
  const _Float16* W = WeffT + h * HDQA * HDQA;
#pragma unroll
  for (int rt = 0; rt < 2; ++rt) {
    v16h qb0 = load_bt_vec(qp + (q0 + rt * 16) * DQA, DQA);       // d=0..31
    v16h qb1 = load_bt_vec(qp + (q0 + rt * 16) * DQA + 32, DQA);  // d=32..63
#pragma unroll
    for (int m = 0; m < 4; ++m) {  // e-tile
      v8f c = {};
      c = wmma_f16(load_a16_vec(W + (m * 16) * 64, 64), qb0, c);
      c = wmma_f16(load_a16_vec(W + (m * 16) * 64 + 32, 64), qb1, c);
#pragma unroll
      for (int v = 0; v < 8; ++v) c[v] *= SC;  // fold softmax scale here
      store_c_f16_col(qmS[w] + (rt * 16) * 64 + m * 16, 64, c);
    }
  }
  // Persistent S^T B-fragments: qm^T columns = query rows, k = e contiguous.
  v16h sb[2][2];
#pragma unroll
  for (int rt = 0; rt < 2; ++rt) {
    sb[rt][0] = load_bt_vec(qmS[w] + (rt * 16) * 64, 64);
    sb[rt][1] = load_bt_vec(qmS[w] + (rt * 16) * 64 + 32, 64);
  }

  float mrow[2] = {-1e30f, -1e30f};
  float lrow[2] = {0.f, 0.f};
  v8f o[2][4];
  v8f zero = {};
#pragma unroll
  for (int rt = 0; rt < 2; ++rt)
#pragma unroll
    for (int m = 0; m < 4; ++m) o[rt][m] = zero;

  for (int kt = 0; kt < 64; ++kt) {
    const _Float16* kt0 = kp + (kt * 32) * DQA;
    // K A-fragments (rows = keys), shared by both row-tiles
    v16h ka00 = load_a16_vec(kt0, DQA);             // keys 0-15,  e 0-31
    v16h ka01 = load_a16_vec(kt0 + 32, DQA);        // keys 0-15,  e 32-63
    v16h ka10 = load_a16_vec(kt0 + 16 * DQA, DQA);  // keys 16-31, e 0-31
    v16h ka11 = load_a16_vec(kt0 + 16 * DQA + 32, DQA);

#pragma unroll
    for (int rt = 0; rt < 2; ++rt) {
      // S^T: lane = query row (col), regs = keys
      v8f s0 = {}, s1 = {};
      s0 = wmma_f16(ka00, sb[rt][0], s0);
      s0 = wmma_f16(ka01, sb[rt][1], s0);
      s1 = wmma_f16(ka10, sb[rt][0], s1);
      s1 = wmma_f16(ka11, sb[rt][1], s1);

      // per-lane max over 16 regs + one cross-half shuffle
      float mx = fmaxf(s0[0], s1[0]);
#pragma unroll
      for (int v = 1; v < 8; ++v) mx = fmaxf(mx, fmaxf(s0[v], s1[v]));
      mx = fmaxf(mx, __shfl_xor(mx, 16, 32));
      float mnew = fmaxf(mrow[rt], mx);

      v8f p0, p1;
      float rs = 0.f;
#pragma unroll
      for (int v = 0; v < 8; ++v) {
        p0[v] = exp2f(s0[v] - mnew);
        p1[v] = exp2f(s1[v] - mnew);
        rs += p0[v] + p1[v];
      }
      rs += __shfl_xor(rs, 16, 32);
      float fs  = exp2f(mrow[rt] - mnew);
      lrow[rt]  = lrow[rt] * fs + rs;
      mrow[rt]  = mnew;

      // store P row-major [qrow][key] via column-major C store (b128)
      store_c_f16_col(pS[w] + (rt * 16) * 32, 32, p0);       // keys 0-15
      store_c_f16_col(pS[w] + (rt * 16) * 32 + 16, 32, p1);  // keys 16-31

      // rescale output accumulator (per-lane scalar factor)
#pragma unroll
      for (int m = 0; m < 4; ++m)
#pragma unroll
        for (int v = 0; v < 8; ++v) o[rt][m][v] *= fs;
    }

    // O^T += V^T @ P^T : A = vt rows (hd, key-contiguous), B = P columns.
    v16h pb0 = load_bt_vec(pS[w], 32);            // row-tile 0
    v16h pb1 = load_bt_vec(pS[w] + 16 * 32, 32);  // row-tile 1
#pragma unroll
    for (int m = 0; m < 4; ++m) {
      v16h va = load_a16_vec(vtp + (m * 16) * ROWS + kt * 32, ROWS);
      o[0][m] = wmma_f16(va, pb0, o[0][m]);
      o[1][m] = wmma_f16(va, pb1, o[1][m]);
    }

    if (kt + 1 < 64) {
      __builtin_prefetch(kp + ((kt + 1) * 32) * DQA, 0, 1);
      __builtin_prefetch(vtp + (kt + 1) * 32, 0, 1);
    }
  }

  // normalize; O^T tile cols = qrow (lane), rows = hd (regs, contiguous) ->
  // b128 stores to row-major aoh.
#pragma unroll
  for (int rt = 0; rt < 2; ++rt) {
    float inv = 1.0f / lrow[rt];
#pragma unroll
    for (int m = 0; m < 4; ++m) {
      v8f c;
#pragma unroll
      for (int v = 0; v < 8; ++v) c[v] = o[rt][m][v] * inv;
      store_c_f16_col(aoh + (b * SQA + q0 + rt * 16) * DQA + h * HDQA + m * 16,
                      DQA, c);
    }
  }
}

// ---------------------------------------------------------------------------
// Projection: out = attn_out @ w_out_eff + b_out  (f32 output)
// ---------------------------------------------------------------------------
__global__ void __launch_bounds__(128) proj_kernel(
    const _Float16* __restrict__ aoh,    // 4096x512 row-major
    const _Float16* __restrict__ woutT,  // 512x512 [e][d]
    const float* __restrict__ bias, float* __restrict__ out)
{
  const int w    = threadIdx.x >> 5;
  const int lane = threadIdx.x & 31;
  const int cl   = lane & 15;
  const int rb   = (lane >> 4) * 8;
  const int row0 = blockIdx.x * 16;
  const int col  = blockIdx.y * 64 + w * 16;
  v8f c = {};
#pragma unroll 4
  for (int ks = 0; ks < 16; ++ks) {
    v16h a = load_a16_vec(aoh + row0 * DQA + ks * 32, DQA);
    v16h b = load_bt_vec(woutT + col * DQA + ks * 32, DQA);
    c = wmma_f16(a, b, c);
  }
  const float bb = bias[col + cl];
#pragma unroll
  for (int v = 0; v < 8; ++v)
    out[(row0 + rb + v) * DQA + col + cl] = c[v] + bb;
}

// ---------------------------------------------------------------------------
// Host-side orchestration
// ---------------------------------------------------------------------------
extern "C" void kernel_launch(void* const* d_in, const int* in_sizes, int n_in,
                              void* d_out, int out_size, void* d_ws, size_t ws_size,
                              hipStream_t stream) {
  const float* x    = (const float*)d_in[0];
  const float* supw = (const float*)d_in[16];
  const float* intf = (const float*)d_in[17];
  const float* wout = (const float*)d_in[18];
  const float* bout = (const float*)d_in[19];

  _Float16* W = (_Float16*)d_ws;
  size_t off = 0;
  _Float16* xh = W + off; off += (size_t)ROWS * DQA;  // x in f16
  _Float16* qkv[3];                                    // q,k row-major; v transposed
  for (int l = 0; l < 3; ++l) { qkv[l] = W + off; off += (size_t)ROWS * DQA; }
  _Float16* aoh = W + off; off += (size_t)ROWS * DQA;
  _Float16* encT[3];
  for (int l = 0; l < 3; ++l) { encT[l] = W + off; off += DQA * QQA; }
  _Float16* measT[3];
  for (int l = 0; l < 3; ++l) { measT[l] = W + off; off += QQA * DQA; }
  _Float16* GRT[3];
  for (int l = 0; l < 3; ++l) { GRT[l] = W + off; off += QQA * QQA; }
  _Float16* GIT[3];
  for (int l = 0; l < 3; ++l) { GIT[l] = W + off; off += QQA * QQA; }
  _Float16* intfT = W + off; off += HQA * HDQA * HDQA;
  _Float16* woutT = W + off; off += DQA * DQA;

  // ---- prep: convert x, transpose-convert all weights, fold scalings ----
  cvt_f16_kernel<<<(ROWS * DQA + 255) / 256, 256, 0, stream>>>(x, xh, ROWS * DQA);
  for (int l = 0; l < 3; ++l) {
    const float* enc  = (const float*)d_in[1 + 5 * l];
    const float* gr   = (const float*)d_in[2 + 5 * l];
    const float* gi   = (const float*)d_in[3 + 5 * l];
    const float* ent  = (const float*)d_in[4 + 5 * l];
    const float* meas = (const float*)d_in[5 + 5 * l];
    cvt_t_kernel<<<(DQA * QQA + 255) / 256, 256, 0, stream>>>(enc, encT[l], DQA, QQA);
    cvt_t_kernel<<<(QQA * DQA + 255) / 256, 256, 0, stream>>>(meas, measT[l], QQA, DQA);
    gemm64_t_kernel<<<16, 256, 0, stream>>>(gr, ent, GRT[l]);
    gemm64_t_kernel<<<16, 256, 0, stream>>>(gi, ent, GIT[l]);
  }
  fold_interf_t_kernel<<<(HQA * HDQA * HDQA + 255) / 256, 256, 0, stream>>>(
      intf, supw, intfT);
  fold_wout_t_kernel<<<(DQA * DQA + 255) / 256, 256, 0, stream>>>(wout, supw, woutT);

  // ---- QNL: q, k row-major; v transposed for PV A-fragments ----
  qnl_kernel<false><<<ROWS / 16, 128, 0, stream>>>(xh, encT[0], GRT[0], GIT[0],
                                                   measT[0], qkv[0]);
  qnl_kernel<false><<<ROWS / 16, 128, 0, stream>>>(xh, encT[1], GRT[1], GIT[1],
                                                   measT[1], qkv[1]);
  qnl_kernel<true><<<ROWS / 16, 128, 0, stream>>>(xh, encT[2], GRT[2], GIT[2],
                                                  measT[2], qkv[2]);

  // ---- attention: 128 query rows per block (32 per wave) ----
  attn_kernel<<<dim3(SQA / 128, BQA * HQA), 128, 0, stream>>>(
      qkv[0], qkv[1], qkv[2], intfT, aoh);

  // ---- projection ----
  proj_kernel<<<dim3(ROWS / 16, DQA / 64), 128, 0, stream>>>(
      aoh, woutT, bout, (float*)d_out);
}
